// Graph_constructor_37924561224035
// MI455X (gfx1250) — matrix operand order repaired and checked by reading, not compiled, not run on hardware
//
#include <hip/hip_runtime.h>
#include <hip/hip_bf16.h>

typedef __attribute__((ext_vector_type(16))) _Float16 v16h;
typedef __attribute__((ext_vector_type(8)))  float    v8f;

#define NV    3000
#define NPAD  3008
#define DG    64
#define NADJ  7
#define NBATCH 8
#define NBINS 4096
#define K_TOP 4500000LL

// ---------------------------------------------------------------- zero hist
__global__ void zero_u32_kernel(unsigned* __restrict__ p, int n) {
    int i = blockIdx.x * 256 + threadIdx.x;
    if (i < n) p[i] = 0u;
}

// ------------------------------------------------- nv = tanh(emb @ W^T + b) -> f16
// grid (NPAD, NADJ), block 64 (thread = output column c)
__global__ void nv_kernel(const float* __restrict__ emb, const float* __restrict__ W,
                          const float* __restrict__ bias, _Float16* __restrict__ out) {
    const int a = blockIdx.y;
    const int row = blockIdx.x;
    const int c = threadIdx.x;
    float val = 0.0f;
    if (row < NV) {
        const float* e = emb + ((size_t)a * NV + row) * DG;
        const float* w = W + ((size_t)a * DG + c) * DG;
        float s = bias[a * DG + c];
        #pragma unroll 8
        for (int k = 0; k < DG; ++k) s += e[k] * w[k];
        val = tanhf(s);
    }
    out[((size_t)a * NPAD + row) * DG + c] = (_Float16)val;
}

// --------------------------------- fused WMMA GEMM + blend + tanh/relu + histogram
// grid (6, 188, NADJ), block 256 (8 waves); wave computes 16(M) x 64(N)
__global__ __launch_bounds__(256) void gemm_adj_kernel(
    const _Float16* __restrict__ nv1h, const _Float16* __restrict__ nv2h,
    const float* __restrict__ cossim, const int* __restrict__ epoch,
    float* __restrict__ adj, unsigned* __restrict__ hist) {

    __shared__ unsigned lhist[NBINS];
    const int tid = threadIdx.x;
    for (int i = tid; i < NBINS; i += 256) lhist[i] = 0u;
    __syncthreads();

    const int a     = blockIdx.z;
    const int mbase = blockIdx.y * 16;
    const int wave  = tid >> 5;
    const int lane  = tid & 31;
    const int hi    = lane >> 4;   // 0 or 1
    const int lo    = lane & 15;
    const int cbase = blockIdx.x * 512 + wave * 64;

    const float prop = fminf((float)epoch[0] * 0.2f, 0.9f);
    const float c1   = 1.0f - prop;

    if (cbase < NPAD) {            // wave-uniform: EXEC stays all-ones for WMMA
        const _Float16* A = nv1h + (size_t)a * NPAD * DG;
        const _Float16* B = nv2h + (size_t)a * NPAD * DG;

        // A fragments (16x32 f16 layout): lane<16 -> K{0..7,16..23}, lane>=16 -> K{8..15,24..31}
        v16h a0, a1;
        {
            const _Float16* p = A + (size_t)(mbase + lo) * DG + hi * 8;
            float4* f0 = reinterpret_cast<float4*>(&a0);
            float4* f1 = reinterpret_cast<float4*>(&a1);
            f0[0] = *(const float4*)(p);        // K 0..7   (or 8..15)
            f0[1] = *(const float4*)(p + 16);   // K 16..23 (or 24..31)
            f1[0] = *(const float4*)(p + 32);   // K 32..39 (or 40..47)
            f1[1] = *(const float4*)(p + 48);   // K 48..55 (or 56..63)
        }

        float* adjbase = adj + (size_t)a * NV * NPAD;

        for (int nt = 0; nt < 4; ++nt) {
            const int ntile = cbase + nt * 16;
            // B fragments (32x16): lane<16 -> K{0..7,16..23}, lane>=16 -> K{8..15,24..31} of nv2 row (col)
            v16h b0, b1;
            {
                const _Float16* q = B + (size_t)(ntile + lo) * DG + hi * 8;
                float4* g0 = reinterpret_cast<float4*>(&b0);
                float4* g1 = reinterpret_cast<float4*>(&b1);
                g0[0] = *(const float4*)(q);
                g0[1] = *(const float4*)(q + 16);
                g1[0] = *(const float4*)(q + 32);
                g1[1] = *(const float4*)(q + 48);
            }
            v8f c = {};
            c = __builtin_amdgcn_wmma_f32_16x16x32_f16(false, a0, false, b0, (short)0, c, false, false);
            c = __builtin_amdgcn_wmma_f32_16x16x32_f16(false, a1, false, b1, (short)0, c, false, false);

            // epilogue: D layout -> element (M = r + 8*hi, N = lo) in c[r]
            #pragma unroll
            for (int r = 0; r < 8; ++r) {
                const int row = mbase + r + 8 * hi;
                const int col = ntile + lo;
                if (row < NV) {
                    float adjv = 0.0f;
                    if (col < NV) {
                        const float av = c1 * cossim[(size_t)row * NV + col] + prop * c[r];
                        const float t = tanhf(av);
                        adjv = t > 0.0f ? t : 0.0f;
                        if (row == col) adjv = 0.0f;
                        if (adjv > 0.0f) {
                            int bin = (int)(adjv * (float)NBINS);
                            if (bin > NBINS - 1) bin = NBINS - 1;
                            atomicAdd(&lhist[bin], 1u);
                        }
                    }
                    adjbase[(size_t)row * NPAD + col] = adjv;
                }
            }
        }
    }
    __syncthreads();
    for (int i = tid; i < NBINS; i += 256) {
        unsigned v = lhist[i];
        if (v) atomicAdd(&hist[a * NBINS + i], v);
    }
}

// ------------------------------------------- threshold = k-th largest via histogram
__global__ void thr_kernel(const unsigned* __restrict__ hist, float* __restrict__ thr) {
    const int a = threadIdx.x;
    if (a >= NADJ) return;
    long long cum = 0;
    float t = 0.0f;
    for (int b = NBINS - 1; b >= 0; --b) {
        cum += (long long)hist[a * NBINS + b];
        if (cum >= K_TOP) { t = (float)b * (1.0f / (float)NBINS); break; }
    }
    thr[a] = t;
}

// ------------------------------------------- rowsum[a][i] = 1 + sum_j masked(adj)
// grid (NV, NADJ), block 128
__global__ void rowsum_kernel(const float* __restrict__ adj, const float* __restrict__ thr,
                              float* __restrict__ rowsum) {
    const int a = blockIdx.y, i = blockIdx.x, tid = threadIdx.x;
    const float t = thr[a];
    const float4* rp = (const float4*)(adj + ((size_t)a * NV + i) * NPAD);
    float s = 0.0f;
    for (int j = tid; j < NPAD / 4; j += 128) {
        const float4 v = rp[j];
        s += (v.x >= t ? v.x : 0.0f) + (v.y >= t ? v.y : 0.0f)
           + (v.z >= t ? v.z : 0.0f) + (v.w >= t ? v.w : 0.0f);
    }
    __shared__ float red[128];
    red[tid] = s;
    __syncthreads();
    for (int st = 64; st > 0; st >>= 1) {
        if (tid < st) red[tid] += red[tid + st];
        __syncthreads();
    }
    if (tid == 0) rowsum[a * NV + i] = 1.0f + red[0];
}

// -------------------------- out[b] = (mask*adj + I) / rowsum, a = time[b] % 7
// grid (ceil(NV*750/256), NBATCH), block 256 (thread = one float4 of a row)
__global__ void out_kernel(const float* __restrict__ adj, const float* __restrict__ thr,
                           const float* __restrict__ rowsum, const int* __restrict__ tindex,
                           float* __restrict__ out) {
    const int b = blockIdx.y;
    const long long idx = (long long)blockIdx.x * 256 + threadIdx.x;
    if (idx >= (long long)NV * (NV / 4)) return;
    const int i  = (int)(idx / (NV / 4));
    const int j4 = (int)(idx % (NV / 4));
    const int a  = ((tindex[b] % NADJ) + NADJ) % NADJ;
    const float t   = thr[a];
    const float inv = 1.0f / rowsum[a * NV + i];
    float4 v = *(const float4*)(adj + ((size_t)a * NV + i) * NPAD + (size_t)j4 * 4);
    const int j0 = j4 * 4;
    float e[4] = {v.x, v.y, v.z, v.w};
    #pragma unroll
    for (int c = 0; c < 4; ++c) {
        float x = e[c] >= t ? e[c] : 0.0f;
        if (j0 + c == i) x = 1.0f;          // diagonal (stored 0, +eye)
        e[c] = x * inv;
    }
    float4 o = make_float4(e[0], e[1], e[2], e[3]);
    *(float4*)(out + ((size_t)b * NV + i) * NV + (size_t)j4 * 4) = o;
}

extern "C" void kernel_launch(void* const* d_in, const int* in_sizes, int n_in,
                              void* d_out, int out_size, void* d_ws, size_t ws_size,
                              hipStream_t stream) {
    (void)in_sizes; (void)n_in; (void)out_size; (void)ws_size;

    const float* emb1   = (const float*)d_in[0];
    const float* emb2   = (const float*)d_in[1];
    const float* W1     = (const float*)d_in[2];
    const float* b1     = (const float*)d_in[3];
    const float* W2     = (const float*)d_in[4];
    const float* b2     = (const float*)d_in[5];
    const float* cossim = (const float*)d_in[6];
    const int*   tindex = (const int*)d_in[7];
    const int*   epoch  = (const int*)d_in[8];
    float*       out    = (float*)d_out;

    // workspace layout (256B aligned)
    char* ws = (char*)d_ws;
    const size_t ADJ_BYTES = (size_t)NADJ * NV * NPAD * sizeof(float);      // 252,672,000
    const size_t NV_BYTES  = (size_t)NADJ * NPAD * DG * sizeof(_Float16);   // 2,695,168
    const size_t HIST_BYTES = (size_t)NADJ * NBINS * sizeof(unsigned);      // 114,688
    float*     adj    = (float*)ws;
    _Float16*  nv1h   = (_Float16*)(ws + ADJ_BYTES);
    _Float16*  nv2h   = (_Float16*)(ws + ADJ_BYTES + NV_BYTES);
    unsigned*  hist   = (unsigned*)(ws + ADJ_BYTES + 2 * NV_BYTES);
    float*     thr    = (float*)(ws + ADJ_BYTES + 2 * NV_BYTES + HIST_BYTES);
    float*     rowsum = (float*)(ws + ADJ_BYTES + 2 * NV_BYTES + HIST_BYTES + 256);

    // 1) zero histograms (harness does not re-zero between replays)
    zero_u32_kernel<<<(NADJ * NBINS + 255) / 256, 256, 0, stream>>>(hist, NADJ * NBINS);

    // 2) nv1 = tanh(emb1 @ W1^T + b1), nv2 = tanh(emb2 @ W2^T + b2)  -> f16, zero-padded rows
    nv_kernel<<<dim3(NPAD, NADJ), 64, 0, stream>>>(emb1, W1, b1, nv1h);
    nv_kernel<<<dim3(NPAD, NADJ), 64, 0, stream>>>(emb2, W2, b2, nv2h);

    // 3) fused WMMA GEMM + blend + tanh/relu + diagonal-zero + store + histogram
    gemm_adj_kernel<<<dim3(6, NPAD / 16, NADJ), 256, 0, stream>>>(nv1h, nv2h, cossim, epoch, adj, hist);

    // 4) per-adjacency top-k threshold from histogram
    thr_kernel<<<1, 32, 0, stream>>>(hist, thr);

    // 5) row sums of masked adj (+1 for identity)
    rowsum_kernel<<<dim3(NV, NADJ), 128, 0, stream>>>(adj, thr, rowsum);

    // 6) batched masked + normalized output
    out_kernel<<<dim3((NV * (NV / 4) + 255) / 256, NBATCH), 256, 0, stream>>>(adj, thr, rowsum, tindex, out);
}
